// LTContextAttention_24068996727079
// MI455X (gfx1250) — matrix operand order repaired and checked by reading, not compiled, not run on hardware
//
#include <hip/hip_runtime.h>
#include <hip/hip_bf16.h>

// LTContextAttention for MI455X (gfx1250, wave32, WMMA bf16 16x16x32).
//
// Shapes: b=8, d=1024, t=8192, dk=512, H=8 (dh=64), G=128 (npat=64).
// Intermediates are t-major ([b][t][dk] bf16) so staging/epilogues move 16-byte
// chunks. Data movers exercised: TDM (TENSOR_LOAD_TO_LDS, TENSORcnt) for the
// attention Q tile, GLOBAL_LOAD_ASYNC_TO_LDS_B128 (ASYNCcnt) for K / out-proj
// B tiles, regular VMEM+DS for everything needing fp32->bf16 conversion.
//
// Pipeline:
//   1) qkv_gemm : [1536,1024] x [1024, 8x8192] bf16-WMMA GEMM -> Qt,Kt,Vt
//   2) attn     : per (b,g,h) 64x64 block: S=Q^T K, masked softmax, O=V P^T,
//                 exact GELU; output aliases Qt (same element set).
//   3) out_gemm : [1024,512] x [512, 8x8192] bf16-WMMA GEMM -> fp32 out*mask
//
// Workspace: 3 x (8*8192*512) bf16 = 192 MiB.

typedef __attribute__((ext_vector_type(16))) __bf16 v16bf;
typedef __attribute__((ext_vector_type(8)))  float  v8f;
typedef int v4i_vs __attribute__((vector_size(16)));  // matches async builtin param
typedef unsigned int u32x4 __attribute__((ext_vector_type(4)));
typedef int i32x4 __attribute__((ext_vector_type(4)));
typedef int i32x8 __attribute__((ext_vector_type(8)));

union FragBf { v16bf v; uint4 q[2]; __bf16 h[16]; };
union Pack8  { uint4 q; __bf16 h[8]; };
union RegTile { float4 v4[4]; float f[16]; };

#if defined(__has_builtin)
#if __has_builtin(__builtin_amdgcn_global_load_async_to_lds_b128)
#define HAVE_ASYNC_LDS 1
#endif
#if __has_builtin(__builtin_amdgcn_tensor_load_to_lds)
#define HAVE_TDM 1
#endif
#endif
#ifndef HAVE_ASYNC_LDS
#define HAVE_ASYNC_LDS 0
#endif
#ifndef HAVE_TDM
#define HAVE_TDM 0
#endif

// Copy 16 bytes global -> LDS. Async (ASYNCcnt-tracked) when available.
__device__ __forceinline__ void copy16B_to_lds(const void* g, void* l) {
#if HAVE_ASYNC_LDS
    __builtin_amdgcn_global_load_async_to_lds_b128(
        (__attribute__((address_space(1))) v4i_vs*)(unsigned long long)g,
        (__attribute__((address_space(3))) v4i_vs*)(unsigned long long)l,
        0, 0);
#else
    *(uint4*)l = *(const uint4*)g;
#endif
}

__device__ __forceinline__ void async_fence() {
#if HAVE_ASYNC_LDS
#if __has_builtin(__builtin_amdgcn_s_wait_asynccnt)
    __builtin_amdgcn_s_wait_asynccnt(0);
#else
    asm volatile("s_wait_asynccnt 0x0" ::: "memory");
#endif
#endif
}

// TDM: load a 64-row x 128-byte tile (row stride 65536 bf16 elements) into LDS
// with a 16-byte pad every 128 bytes (-> 144 B = 72-element LDS row stride).
// D# bitfields per CDNA5 ISA.
__device__ __forceinline__ void tdm_load_tile_64x64(const __bf16* gsrc, __bf16* lds) {
#if HAVE_TDM
    const unsigned long long ga = (unsigned long long)gsrc;
    const unsigned int la = (unsigned int)(unsigned long long)lds;
    u32x4 g0;
    g0[0] = 1u;                                       // count=1, user D#
    g0[1] = la;                                       // lds_addr (bytes)
    g0[2] = (unsigned int)(ga & 0xFFFFFFFFull);       // global_addr[31:0]
    g0[3] = (unsigned int)((ga >> 32) & 0x01FFFFFFull)
          | (2u << 30);                               // global_addr[56:32] | type=2
    i32x8 g1;
    g1[0] = (1 << 16)      // data_size = 2 bytes
          | (1 << 20)      // pad_enable
          | (4 << 22)      // pad_interval: 32 DWORDs (=128 B) before padding
          | (3 << 25);     // pad_amount: 4 DWORDs (=16 B)
    g1[1] = 0;             // atomic_barrier off; tensor_dim0[15:0] = 0
    g1[2] = 1;             // tensor_dim0 = 0x10000; tensor_dim1[15:0] = 0
    g1[3] = 16 | (64 << 16);   // tensor_dim1 = 0x100000; tile_dim0 = 64
    g1[4] = 64;            // tile_dim1 = 64, tile_dim2 = 0
    g1[5] = 65536;         // tensor_dim0_stride (elements) low 32
    g1[6] = 0;             // stride hi / dim1_stride lo
    g1[7] = 0;
    const i32x4 z4 = {0, 0, 0, 0};
    const i32x8 z8 = {0, 0, 0, 0, 0, 0, 0, 0};
    __builtin_amdgcn_tensor_load_to_lds(g0, g1, z4, z4, z8, 0);
#else
    (void)gsrc; (void)lds;
#endif
}

__device__ __forceinline__ void tdm_wait() {
#if HAVE_TDM
#if __has_builtin(__builtin_amdgcn_s_wait_tensorcnt)
    __builtin_amdgcn_s_wait_tensorcnt(0);
#else
    asm volatile("s_wait_tensorcnt 0x0" ::: "memory");
#endif
#endif
}

// A-matrix fragment (16x32 bf16, M = lane%16):
//   elements 0..7 = K 8*hi..8*hi+7 ; elements 8..15 = K 16+8*hi..16+8*hi+7
__device__ __forceinline__ v16bf load_frag_a(const __bf16* rowbase, int stride, int lane) {
    const int lo = lane & 15, hi = lane >> 4;
    const uint4* p = (const uint4*)(rowbase + lo * stride);
    FragBf f;
    f.q[0] = p[hi];
    f.q[1] = p[2 + hi];
    return f.v;
}

// B-matrix fragment (32x16 bf16, N = lane%16):
//   elements 0..7 = K 8*hi..8*hi+7 ; elements 8..15 = K 16+8*hi..16+8*hi+7
__device__ __forceinline__ v16bf load_frag_b(const __bf16* rowbase, int stride, int lane) {
    const int lo = lane & 15, hi = lane >> 4;
    const uint4* p = (const uint4*)(rowbase + lo * stride);
    FragBf f;
    f.q[0] = p[hi];
    f.q[1] = p[2 + hi];
    return f.v;
}

__device__ __forceinline__ v8f zero8() {
    v8f z = {0.f, 0.f, 0.f, 0.f, 0.f, 0.f, 0.f, 0.f};
    return z;
}

// ---------------------------------------------------------------------------
// Kernel 1: QKV projection GEMM, double-buffered LDS.
// C[1536, 65536] = [Wq;Wk;Wv] x qk (both fp32 -> bf16 on the staging path).
// Output t-major: Qt/Kt/Vt[b][t][512] bf16 (packed b128 epilogue stores).
// ---------------------------------------------------------------------------
__global__ __launch_bounds__(256) void qkv_gemm(
    const float* __restrict__ qk,
    const float* __restrict__ wq, const float* __restrict__ bq,
    const float* __restrict__ wk, const float* __restrict__ bk,
    const float* __restrict__ wv, const float* __restrict__ bv,
    __bf16* __restrict__ qt_out, __bf16* __restrict__ kt_out,
    __bf16* __restrict__ vt_out)
{
    __shared__ __align__(16) __bf16 lA[2][128 * 40];  // [row][k], stride 40
    __shared__ __align__(16) __bf16 lB[2][128 * 40];  // [n][k]   (X transposed)

    const int tid  = threadIdx.x;
    const int lane = tid & 31;
    const int wave = tid >> 5;
    const int wm = wave & 1, wn = wave >> 1;

    const int m0 = blockIdx.y * 128;            // stacked W rows
    const int b  = blockIdx.x >> 6;
    const int t0 = (blockIdx.x & 63) * 128;
    const float* X = qk + (size_t)b * 1024 * 8192;

    const int lr = tid >> 1;          // tile row / col staged by this thread
    const int kh = (tid & 1) * 16;    // K half

    const int R = m0 + lr;
    const float* wsrc = (R < 512) ? wq : (R < 1024 ? wk : wv);
    const float* arow = wsrc + (size_t)(R & 511) * 1024 + kh;
    const float* brow = X + (size_t)kh * 8192 + t0 + lr;

    v8f acc[4][2];
    #pragma unroll
    for (int i = 0; i < 4; ++i)
        #pragma unroll
        for (int j = 0; j < 2; ++j) acc[i][j] = zero8();

    RegTile ra, rb;
    auto ld = [&](int k0) {
        const float4* pa = (const float4*)(arow + k0);
        #pragma unroll
        for (int j = 0; j < 4; ++j) ra.v4[j] = pa[j];
        const float* pb = brow + (size_t)k0 * 8192;
        #pragma unroll
        for (int j = 0; j < 16; ++j) rb.f[j] = pb[(size_t)j * 8192];
        __builtin_prefetch(pb + (size_t)32 * 8192, 0, 1);
    };
    auto st = [&](int buf) {
        FragBf ta, tb;
        #pragma unroll
        for (int j = 0; j < 16; ++j) {
            ta.h[j] = (__bf16)ra.f[j];
            tb.h[j] = (__bf16)rb.f[j];
        }
        uint4* da = (uint4*)&lA[buf][lr * 40 + kh];
        da[0] = ta.q[0]; da[1] = ta.q[1];
        uint4* db = (uint4*)&lB[buf][lr * 40 + kh];
        db[0] = tb.q[0]; db[1] = tb.q[1];
    };

    ld(0);
    int buf = 0;
    for (int k0 = 0; k0 < 1024; k0 += 32) {
        st(buf);
        __syncthreads();
        if (k0 + 32 < 1024) ld(k0 + 32);

        v16bf afr[4], bfr[2];
        #pragma unroll
        for (int mt = 0; mt < 4; ++mt)
            afr[mt] = load_frag_a(&lA[buf][(wm * 64 + mt * 16) * 40], 40, lane);
        #pragma unroll
        for (int nt = 0; nt < 2; ++nt)
            bfr[nt] = load_frag_b(&lB[buf][(wn * 32 + nt * 16) * 40], 40, lane);
        #pragma unroll
        for (int mt = 0; mt < 4; ++mt)
            #pragma unroll
            for (int nt = 0; nt < 2; ++nt)
                acc[mt][nt] = __builtin_amdgcn_wmma_f32_16x16x32_bf16(
                    false, afr[mt], false, bfr[nt], (short)0, acc[mt][nt], false, false);
        buf ^= 1;
    }

    // Epilogue: rows are consecutive dk per lane -> packed b128 t-major stores.
    const int lo = lane & 15, hi = lane >> 4;
    #pragma unroll
    for (int mt = 0; mt < 4; ++mt) {
        #pragma unroll
        for (int nt = 0; nt < 2; ++nt) {
            const int Rbase = m0 + wm * 64 + mt * 16 + 8 * hi;  // + v
            const int c     = t0 + wn * 32 + nt * 16 + lo;
            const int sel   = Rbase >> 9;
            const int r     = Rbase & 511;
            const float* bias = (sel == 0) ? bq : (sel == 1 ? bk : bv);
            __bf16* dstT = (sel == 0) ? qt_out : (sel == 1 ? kt_out : vt_out);
            Pack8 pk;
            #pragma unroll
            for (int v = 0; v < 8; ++v)
                pk.h[v] = (__bf16)(acc[mt][nt][v] + bias[r + v]);
            *(uint4*)&dstT[((size_t)b * 8192 + c) * 512 + r] = pk.q;
        }
    }
}

// ---------------------------------------------------------------------------
// Kernel 2: attention per (b, g, h). 128 threads = 4 waves.
// Q tile: one TDM descriptor (TENSORcnt). K tile: async b128 (ASYNCcnt).
// V tile: b128 loads + b16 LDS scatter transpose to [d][j].
// ---------------------------------------------------------------------------
__global__ __launch_bounds__(128) void attn_kernel(
    const __bf16* __restrict__ qt_g, const __bf16* __restrict__ kt_g,
    const __bf16* __restrict__ vt_g, const float* __restrict__ masks,
    __bf16* __restrict__ act)
{
    __shared__ __align__(16) __bf16 qt[64 * 72];  // [i(p)][d]
    __shared__ __align__(16) __bf16 kt[64 * 72];  // [j(p)][d]
    __shared__ __align__(16) __bf16 vt[64 * 72];  // [d][j]
    __shared__ __align__(16) __bf16 pt[64 * 72];  // [i][j]

    const int tid  = threadIdx.x;
    const int lane = tid & 31;
    const int w    = tid >> 5;
    const int lo = lane & 15, hi = lane >> 4;

    const int bg = blockIdx.x >> 3;
    const int h  = blockIdx.x & 7;
    const int b  = bg >> 7;
    const int g  = bg & 127;

    const size_t rowbase = (size_t)b * 8192 + g;    // t-index base (t = g + 128p)
    const size_t tile0   = rowbase * 512 + h * 64;  // element offset of (p=0,d=0)

#if HAVE_TDM
    if (w == 0) tdm_load_tile_64x64(qt_g + tile0, qt);  // wave-uniform branch
#endif

    // Stage K (async) and V (transpose): 64 rows x 8 chunks of 8 bf16.
    #pragma unroll
    for (int s = 0; s < 4; ++s) {
        const int idx = tid + 128 * s;
        const int p   = idx >> 3;
        const int ch  = (idx & 7) * 8;
        const size_t goff = tile0 + (size_t)(128 * p) * 512 + ch;
#if !HAVE_TDM
        copy16B_to_lds(qt_g + goff, &qt[p * 72 + ch]);
#endif
        copy16B_to_lds(kt_g + goff, &kt[p * 72 + ch]);
        Pack8 rv;
        rv.q = *(const uint4*)(vt_g + goff);
        #pragma unroll
        for (int j = 0; j < 8; ++j) vt[(ch + j) * 72 + p] = rv.h[j];
    }
#if HAVE_TDM
    if (w == 0) tdm_wait();
#endif
    async_fence();
    __syncthreads();

    // S rows i in [16w,16w+16), all 64 j; K over dh=64 (2 WMMA steps).
    v8f accS[4];
    #pragma unroll
    for (int nt = 0; nt < 4; ++nt) accS[nt] = zero8();
    #pragma unroll
    for (int ks = 0; ks < 2; ++ks) {
        const v16bf a = load_frag_a(&qt[(w * 16) * 72 + ks * 32], 72, lane);
        #pragma unroll
        for (int nt = 0; nt < 4; ++nt) {
            const v16bf bf = load_frag_b(&kt[(nt * 16) * 72 + ks * 32], 72, lane);
            accS[nt] = __builtin_amdgcn_wmma_f32_16x16x32_bf16(
                false, a, false, bf, (short)0, accS[nt], false, false);
        }
    }

    // Scale + key-side mask (column j -> time g + 128*j).
    float mk[4];
    #pragma unroll
    for (int nt = 0; nt < 4; ++nt)
        mk[nt] = masks[(size_t)b * 8192 + g + 128 * (nt * 16 + lo)];
    #pragma unroll
    for (int nt = 0; nt < 4; ++nt)
        #pragma unroll
        for (int v = 0; v < 8; ++v) {
            const float x = accS[nt][v] * 0.125f;  // 1/sqrt(64)
            accS[nt][v] = (mk[nt] > 0.f) ? x : -1e9f;
        }

    // Softmax over j (row i = fixed (v,hi); reduce across 16 lanes of a half).
    #pragma unroll
    for (int v = 0; v < 8; ++v) {
        float m = accS[0][v];
        #pragma unroll
        for (int nt = 1; nt < 4; ++nt) m = fmaxf(m, accS[nt][v]);
        #pragma unroll
        for (int xm = 1; xm <= 8; xm <<= 1) m = fmaxf(m, __shfl_xor(m, xm, 32));
        float s = 0.f;
        #pragma unroll
        for (int nt = 0; nt < 4; ++nt) {
            const float e = __expf(accS[nt][v] - m);
            accS[nt][v] = e;
            s += e;
        }
        #pragma unroll
        for (int xm = 1; xm <= 8; xm <<= 1) s += __shfl_xor(s, xm, 32);
        const float inv = 1.0f / s;
        #pragma unroll
        for (int nt = 0; nt < 4; ++nt) accS[nt][v] *= inv;
    }

    // P -> LDS [i][j] bf16.
    #pragma unroll
    for (int nt = 0; nt < 4; ++nt)
        #pragma unroll
        for (int v = 0; v < 8; ++v)
            pt[(w * 16 + v + 8 * hi) * 72 + nt * 16 + lo] = (__bf16)accS[nt][v];
    __syncthreads();

    // O[d,i] = sum_j V[d,j] P[i,j]; wave handles i in [16w,16w+16).
    v8f accO[4];
    #pragma unroll
    for (int mt = 0; mt < 4; ++mt) accO[mt] = zero8();
    #pragma unroll
    for (int ks = 0; ks < 2; ++ks) {
        const v16bf bf = load_frag_b(&pt[(w * 16) * 72 + ks * 32], 72, lane);
        #pragma unroll
        for (int mt = 0; mt < 4; ++mt) {
            const v16bf a = load_frag_a(&vt[(mt * 16) * 72 + ks * 32], 72, lane);
            accO[mt] = __builtin_amdgcn_wmma_f32_16x16x32_bf16(
                false, a, false, bf, (short)0, accO[mt], false, false);
        }
    }

    // Exact GELU, packed b128 t-major stores (aliases Qt: same element set).
    #pragma unroll
    for (int mt = 0; mt < 4; ++mt) {
        Pack8 pk;
        #pragma unroll
        for (int v = 0; v < 8; ++v) {
            const float x = accO[mt][v];
            pk.h[v] = (__bf16)(0.5f * x * (1.0f + erff(x * 0.70710678118654752f)));
        }
        const int i = w * 16 + lo;
        const int d = mt * 16 + 8 * hi;
        *(uint4*)&act[(rowbase + 128 * i) * 512 + h * 64 + d] = pk.q;
    }
}

// ---------------------------------------------------------------------------
// Kernel 3: output projection GEMM, double-buffered; B tile streams via
// async b128 into LDS (act is t-major: K contiguous). fp32 out with bias+mask.
// ---------------------------------------------------------------------------
__global__ __launch_bounds__(256) void out_gemm(
    const __bf16* __restrict__ actT, const float* __restrict__ wo,
    const float* __restrict__ bo, const float* __restrict__ masks,
    float* __restrict__ out)
{
    __shared__ __align__(16) __bf16 lA[2][128 * 40];
    __shared__ __align__(16) __bf16 lB[2][128 * 40];

    const int tid  = threadIdx.x;
    const int lane = tid & 31;
    const int wave = tid >> 5;
    const int wm = wave & 1, wn = wave >> 1;

    const int m0 = blockIdx.y * 128;
    const int b  = blockIdx.x >> 6;
    const int t0 = (blockIdx.x & 63) * 128;

    const int lr = tid >> 1;
    const int kh = (tid & 1) * 16;

    const float*  arow = wo + (size_t)(m0 + lr) * 512 + kh;
    const __bf16* brow = actT + ((size_t)b * 8192 + t0 + lr) * 512 + kh;

    v8f acc[4][2];
    #pragma unroll
    for (int i = 0; i < 4; ++i)
        #pragma unroll
        for (int j = 0; j < 2; ++j) acc[i][j] = zero8();

    RegTile ra;
    auto ldA = [&](int k0) {
        const float4* pa = (const float4*)(arow + k0);
        #pragma unroll
        for (int j = 0; j < 4; ++j) ra.v4[j] = pa[j];
    };
    auto stA = [&](int buf) {
        FragBf ta;
        #pragma unroll
        for (int j = 0; j < 16; ++j) ta.h[j] = (__bf16)ra.f[j];
        uint4* da = (uint4*)&lA[buf][lr * 40 + kh];
        da[0] = ta.q[0]; da[1] = ta.q[1];
    };
    auto issueB = [&](int k0, int buf) {
        copy16B_to_lds(brow + k0,     &lB[buf][lr * 40 + kh]);
        copy16B_to_lds(brow + k0 + 8, &lB[buf][lr * 40 + kh + 8]);
    };

    ldA(0);
    issueB(0, 0);
    int buf = 0;
    for (int k0 = 0; k0 < 512; k0 += 32) {
        stA(buf);
        async_fence();
        __syncthreads();
        if (k0 + 32 < 512) {
            ldA(k0 + 32);
            issueB(k0 + 32, buf ^ 1);
        }

        v16bf afr[4], bfr[2];
        #pragma unroll
        for (int mt = 0; mt < 4; ++mt)
            afr[mt] = load_frag_a(&lA[buf][(wm * 64 + mt * 16) * 40], 40, lane);
        #pragma unroll
        for (int nt = 0; nt < 2; ++nt)
            bfr[nt] = load_frag_b(&lB[buf][(wn * 32 + nt * 16) * 40], 40, lane);
        #pragma unroll
        for (int mt = 0; mt < 4; ++mt)
            #pragma unroll
            for (int nt = 0; nt < 2; ++nt)
                acc[mt][nt] = __builtin_amdgcn_wmma_f32_16x16x32_bf16(
                    false, afr[mt], false, bfr[nt], (short)0, acc[mt][nt], false, false);
        buf ^= 1;
    }

    const int lo = lane & 15, hi = lane >> 4;
    #pragma unroll
    for (int mt = 0; mt < 4; ++mt) {
        #pragma unroll
        for (int nt = 0; nt < 2; ++nt) {
            #pragma unroll
            for (int v = 0; v < 8; ++v) {
                const int Rr = m0 + wm * 64 + mt * 16 + v + 8 * hi;
                const int c  = t0 + wn * 32 + nt * 16 + lo;
                const float msk = masks[(size_t)b * 8192 + c];
                out[((size_t)b * 1024 + Rr) * 8192 + c] =
                    (acc[mt][nt][v] + bo[Rr]) * msk;
            }
        }
    }
}

// ---------------------------------------------------------------------------
extern "C" void kernel_launch(void* const* d_in, const int* in_sizes, int n_in,
                              void* d_out, int out_size, void* d_ws, size_t ws_size,
                              hipStream_t stream) {
    (void)in_sizes; (void)n_in; (void)out_size; (void)ws_size;

    const float* qk    = (const float*)d_in[0];
    const float* masks = (const float*)d_in[1];
    const float* wq    = (const float*)d_in[2];
    const float* bq    = (const float*)d_in[3];
    const float* wk    = (const float*)d_in[4];
    const float* bk    = (const float*)d_in[5];
    const float* wv    = (const float*)d_in[6];
    const float* bv    = (const float*)d_in[7];
    const float* wo    = (const float*)d_in[8];
    const float* bo    = (const float*)d_in[9];
    float* out = (float*)d_out;

    const size_t elems = (size_t)8 * 8192 * 512;  // per Q/K/V tensor (t-major)
    __bf16* qt = (__bf16*)d_ws;
    __bf16* kt = qt + elems;
    __bf16* vt = kt + elems;
    __bf16* act = qt;  // attention output aliases Qt (exact element reuse)

    qkv_gemm<<<dim3(512, 12), 256, 0, stream>>>(qk, wq, bq, wk, bk, wv, bv,
                                                qt, kt, vt);
    attn_kernel<<<dim3(8192), 128, 0, stream>>>(qt, kt, vt, masks, act);
    out_gemm<<<dim3(512, 8), 256, 0, stream>>>(act, wo, bo, masks, out);
}
